// TransformerBlock_26843545600028
// MI455X (gfx1250) — 8-bit matrix operand order repaired and checked by reading, compiled, not run on hardware
//
#include <hip/hip_runtime.h>
#include <hip/hip_bf16.h>
#include <cstdint>


#define BB   4
#define SS_  1024
#define EE   1024
#define HH   16
#define DD   64
#define MLPN 4096
#define NT   (BB * SS_)

typedef __attribute__((ext_vector_type(8))) int v8i;
typedef __attribute__((ext_vector_type(4))) int v4i;

enum {
  SL_X = 0, SL_WQ, SL_WK, SL_WV, SL_WO, SL_W1, SL_W2,
  SL_BQ, SL_BK, SL_BV, SL_BO, SL_B1, SL_B2,
  SL_Q, SL_K, SL_V, SL_SC, SL_AT, SL_CTX, SL_AO, SL_H, SL_M1, SL_M2,
  SL_COUNT
};


__global__ void init_scales_kernel(float* s) {
  if (threadIdx.x < 64) s[threadIdx.x] = 0.0f;
}

__global__ void absmax_kernel(const float* __restrict__ x, long long n,
                              float* out_slot, int mode) {
  __shared__ float red[256];
  float m = 0.0f;
  long long i = (long long)blockIdx.x * blockDim.x + threadIdx.x;
  long long stride = (long long)gridDim.x * blockDim.x;
  for (; i < n; i += stride) {
    float v = x[i];
    v = mode ? fmaxf(v, 0.0f) : fabsf(v);
    m = fmaxf(m, v);
  }
  red[threadIdx.x] = m;
  __syncthreads();
  for (int s2 = 128; s2 > 0; s2 >>= 1) {
    if (threadIdx.x < (unsigned)s2)
      red[threadIdx.x] = fmaxf(red[threadIdx.x], red[threadIdx.x + s2]);
    __syncthreads();
  }
  if (threadIdx.x == 0)
    atomicMax((unsigned int*)out_slot, __float_as_uint(red[0]));
}


__device__ __forceinline__ float fq_scale(float absmax, float premul, float qmax) {
  return fmaxf(absmax * premul / qmax, 1e-8f);
}

__global__ void quant_i8_kernel(const float* __restrict__ in, int8_t* __restrict__ out,
                                const float* __restrict__ scales, int slot,
                                float premul, float lo, float hi, long long n) {
  float s = fq_scale(scales[slot], premul, 127.0f);
  long long i = (long long)blockIdx.x * blockDim.x + threadIdx.x;
  if (i < n)
    out[i] = (int8_t)fminf(fmaxf(rintf(in[i] * premul / s), lo), hi);
}

__global__ void relu_quant_u8_kernel(const float* __restrict__ in,
                                     unsigned char* __restrict__ out,
                                     const float* __restrict__ scales, int slot,
                                     long long n) {
  float s = fq_scale(scales[slot], 1.0f, 255.0f);
  long long i = (long long)blockIdx.x * blockDim.x + threadIdx.x;
  if (i < n)
    out[i] = (unsigned char)fminf(fmaxf(rintf(fmaxf(in[i], 0.0f) / s), 0.0f), 255.0f);
}

__global__ void fq_float_kernel(const float* __restrict__ in, float* __restrict__ out,
                                const float* __restrict__ scales, int slot,
                                float lo, float hi, long long n) {
  float s = fq_scale(scales[slot], 1.0f, 127.0f);
  long long i = (long long)blockIdx.x * blockDim.x + threadIdx.x;
  if (i < n)
    out[i] = fminf(fmaxf(rintf(in[i] / s), lo), hi) * s;
}

__global__ void quant_perm_kernel(const float* __restrict__ in, int8_t* __restrict__ out,
                                  const float* __restrict__ scales, int slot,
                                  float premul, int mode) {
  float s = fq_scale(scales[slot], premul, 127.0f);
  long long i = (long long)blockIdx.x * blockDim.x + threadIdx.x;
  if (i >= (long long)NT * EE) return;
  float q = fminf(fmaxf(rintf(in[i] * premul / s), -128.0f), 127.0f);
  long long o;
  if (mode == 2) {
    int b = (int)(i >> 20);
    int h = (int)((i >> 16) & 15);
    int t = (int)((i >> 6) & 1023);
    int d = (int)(i & 63);
    o = ((long long)(b * SS_ + t) << 10) + (h << 6) + d;
  } else {
    int b = (int)(i >> 20);
    int t = (int)((i >> 10) & 1023);
    int h = (int)((i >> 6) & 15);
    int d = (int)(i & 63);
    long long bh = (long long)(b * HH + h) << 16;
    o = (mode == 0) ? bh + (t << 6) + d
                    : bh + (d << 10) + t;
  }
  out[o] = (int8_t)q;
}


template <bool SA, bool SB>
__global__ __launch_bounds__(256) void gemm_i8_wmma(
    const int8_t* __restrict__ A, const int8_t* __restrict__ Bm,
    float* __restrict__ C, int M, int N, int K,
    long long strideA, long long strideB, long long strideC,
    const float* __restrict__ scales, int slotA, float qmaxA, float premulA,
    int slotB, const float* __restrict__ bias) {
  __shared__ __align__(16) int8_t smB[2][64 * 64];

  const int tid  = threadIdx.x;
  const int lane = tid & 31;
  const int wave = tid >> 5;
  const int row0 = blockIdx.y * 256 + wave * 32;
  const int col0 = blockIdx.x * 64;

  A  += (long long)blockIdx.z * strideA;
  Bm += (long long)blockIdx.z * strideB;
  C  += (long long)blockIdx.z * strideC;

  const int l16  = lane & 15;
  const int half = lane >> 4;

  const int bcol = tid >> 2;
  const int bko  = (tid & 3) << 4;
  const int8_t* bsrc = Bm + (long long)(col0 + bcol) * K + bko;
  const unsigned ldsA = (unsigned)(size_t)(&smB[0][bcol * 64 + bko]);
  const unsigned ldsB = (unsigned)(size_t)(&smB[1][bcol * 64 + bko]);

  {
    unsigned long long ga = (unsigned long long)(size_t)(bsrc);
    asm volatile("global_load_async_to_lds_b128 %0, %1, off"
                 :: "v"(ldsA), "v"(ga) : "memory");
  }

  v8i acc[2][4] = {};
  const int8_t* arow0 = A + (long long)(row0 + l16) * K;
  const int8_t* arow1 = arow0 + (long long)16 * K;

  int buf = 0;
  for (int k0 = 0; k0 < K; k0 += 64) {
    v8i a0, a1;
#pragma unroll
    for (int v = 0; v < 8; ++v) {
      int kA = ((v >> 2) << 5) | (half << 4) | ((v & 3) << 2);
      a0[v] = *(const int*)(arow0 + k0 + kA);
      a1[v] = *(const int*)(arow1 + k0 + kA);
    }

    asm volatile("s_wait_asynccnt 0x0" ::: "memory");
    __syncthreads();

    if (k0 + 64 < K) {
      unsigned long long ga = (unsigned long long)(size_t)(bsrc + k0 + 64);
      asm volatile("global_load_async_to_lds_b128 %0, %1, off"
                   :: "v"(buf ? ldsA : ldsB), "v"(ga) : "memory");
      __builtin_prefetch(arow0 + k0 + 64, 0, 3);
      __builtin_prefetch(arow1 + k0 + 64, 0, 3);
    }

    const int8_t* sb = smB[buf];
    v8i bf[4];
#pragma unroll
    for (int j = 0; j < 4; ++j) {
      const int8_t* bb = sb + (j * 16 + l16) * 64 + half * 16;
      v4i lo = *(const v4i*)(bb);
      v4i hi = *(const v4i*)(bb + 32);
      bf[j][0] = lo[0]; bf[j][1] = lo[1]; bf[j][2] = lo[2]; bf[j][3] = lo[3];
      bf[j][4] = hi[0]; bf[j][5] = hi[1]; bf[j][6] = hi[2]; bf[j][7] = hi[3];
    }
#pragma unroll
    for (int j = 0; j < 4; ++j)
      acc[0][j] = __builtin_amdgcn_wmma_i32_16x16x64_iu8(SA, a0, SB, bf[j],
                                                         acc[0][j], false, false);
#pragma unroll
    for (int j = 0; j < 4; ++j)
      acc[1][j] = __builtin_amdgcn_wmma_i32_16x16x64_iu8(SA, a1, SB, bf[j],
                                                         acc[1][j], false, false);
    buf ^= 1;
  }

  float sA = fmaxf(scales[slotA] * premulA / qmaxA, 1e-8f);
  float sB = fmaxf(scales[slotB] / 127.0f, 1e-8f);
  float sAB = sA * sB;

#pragma unroll
  for (int t = 0; t < 2; ++t) {
#pragma unroll
    for (int j = 0; j < 4; ++j) {
      int col = col0 + j * 16 + l16;
      float bv = bias ? bias[col] : 0.0f;
#pragma unroll
      for (int r = 0; r < 8; ++r) {
        int m = row0 + t * 16 + r + half * 8;
        C[(long long)m * N + col] = (float)acc[t][j][r] * sAB + bv;
      }
    }
  }
}

__global__ void softmax_fq_kernel(float* __restrict__ sc,
                                  const float* __restrict__ scales, int slot) {
  __shared__ float red[256];
  const long long base = (long long)blockIdx.x * 1024;
  const int t = threadIdx.x;
  float s = fmaxf(scales[slot] / 127.0f, 1e-8f);
  float v[4];
  float m = -3.4e38f;
#pragma unroll
  for (int j = 0; j < 4; ++j) {
    float x = sc[base + t + j * 256];
    v[j] = fminf(fmaxf(rintf(x / s), -128.0f), 127.0f) * s;
    m = fmaxf(m, v[j]);
  }
  red[t] = m; __syncthreads();
  for (int s2 = 128; s2 > 0; s2 >>= 1) {
    if (t < s2) red[t] = fmaxf(red[t], red[t + s2]);
    __syncthreads();
  }
  m = red[0]; __syncthreads();
  float sum = 0.0f;
#pragma unroll
  for (int j = 0; j < 4; ++j) { v[j] = expf(v[j] - m); sum += v[j]; }
  red[t] = sum; __syncthreads();
  for (int s2 = 128; s2 > 0; s2 >>= 1) {
    if (t < s2) red[t] += red[t + s2];
    __syncthreads();
  }
  float inv = 1.0f / red[0];
#pragma unroll
  for (int j = 0; j < 4; ++j) sc[base + t + j * 256] = v[j] * inv;
}

__global__ void add_fq_ln_kernel(const float* __restrict__ x,
                                 const float* __restrict__ y,
                                 const float* __restrict__ scales, int sx, int sy,
                                 float* __restrict__ out) {
  __shared__ float red[256];
  const long long base = (long long)blockIdx.x * EE;
  const int t = threadIdx.x;
  float ssx = fmaxf(scales[sx] / 127.0f, 1e-8f);
  float ssy = fmaxf(scales[sy] / 127.0f, 1e-8f);
  float v[4];
  float lsum = 0.0f;
#pragma unroll
  for (int j = 0; j < 4; ++j) {
    long long idx = base + t + j * 256;
    float a = fminf(fmaxf(rintf(x[idx] / ssx), -128.0f), 127.0f) * ssx;
    float b = fminf(fmaxf(rintf(y[idx] / ssy), -128.0f), 127.0f) * ssy;
    v[j] = a + b;
    lsum += v[j];
  }
  red[t] = lsum; __syncthreads();
  for (int s2 = 128; s2 > 0; s2 >>= 1) {
    if (t < s2) red[t] += red[t + s2];
    __syncthreads();
  }
  float mu = red[0] * (1.0f / EE); __syncthreads();
  float lsq = 0.0f;
#pragma unroll
  for (int j = 0; j < 4; ++j) { float d = v[j] - mu; lsq += d * d; }
  red[t] = lsq; __syncthreads();
  for (int s2 = 128; s2 > 0; s2 >>= 1) {
    if (t < s2) red[t] += red[t + s2];
    __syncthreads();
  }
  float rs = rsqrtf(red[0] * (1.0f / EE) + 1e-5f);
#pragma unroll
  for (int j = 0; j < 4; ++j)
    out[base + t + j * 256] = (v[j] - mu) * rs;
}


extern "C" void kernel_launch(void* const* d_in, const int* in_sizes, int n_in,
                              void* d_out, int out_size, void* d_ws, size_t ws_size,
                              hipStream_t stream) {
  const float* x  = (const float*)d_in[0];
  const float* Wq = (const float*)d_in[1];
  const float* bq = (const float*)d_in[2];
  const float* Wk = (const float*)d_in[3];
  const float* bk = (const float*)d_in[4];
  const float* Wv = (const float*)d_in[5];
  const float* bv = (const float*)d_in[6];
  const float* Wo = (const float*)d_in[7];
  const float* bo = (const float*)d_in[8];
  const float* W1 = (const float*)d_in[9];
  const float* b1 = (const float*)d_in[10];
  const float* W2 = (const float*)d_in[11];
  const float* b2 = (const float*)d_in[12];
  float* out = (float*)d_out;

  char* ws = (char*)d_ws;
  size_t off = 0;
  auto take = [&](size_t bytes) -> char* {
    char* p = ws + off;
    off = (off + bytes + 255) & ~(size_t)255;
    return p;
  };

  const long long nTE  = (long long)NT * EE;
  const long long nSc  = (long long)BB * HH * SS_ * SS_;
  const long long nM1  = (long long)NT * MLPN;

  float*  scales = (float*)take(64 * 4);
  float*  bqf = (float*)take(EE * 4);
  float*  bkf = (float*)take(EE * 4);
  float*  bvf = (float*)take(EE * 4);
  float*  bof = (float*)take(EE * 4);
  float*  b1f = (float*)take(MLPN * 4);
  float*  b2f = (float*)take(EE * 4);
  int8_t* qx  = (int8_t*)take(nTE);
  int8_t* qWq = (int8_t*)take((size_t)EE * EE);
  int8_t* qWk = (int8_t*)take((size_t)EE * EE);
  int8_t* qWv = (int8_t*)take((size_t)EE * EE);
  int8_t* qWo = (int8_t*)take((size_t)EE * EE);
  int8_t* qW1 = (int8_t*)take((size_t)MLPN * EE);
  int8_t* qW2 = (int8_t*)take((size_t)EE * MLPN);
  float*  qf  = (float*)take(nTE * 4);
  float*  kf  = (float*)take(nTE * 4);
  float*  vf  = (float*)take(nTE * 4);
  int8_t* q8  = (int8_t*)take(nTE);
  int8_t* k8  = (int8_t*)take(nTE);
  int8_t* v8t = (int8_t*)take(nTE);
  float*  scoresf = (float*)take(nSc * 4);
  int8_t* attn8   = (int8_t*)take(nSc);
  float*  ctxf    = (float*)take(nTE * 4);
  int8_t* ctx8    = (int8_t*)take(nTE);
  float*  aof     = (float*)take(nTE * 4);
  float*  hf      = (float*)take(nTE * 4);
  int8_t* h8      = (int8_t*)take(nTE);
  float*  m1f     = (float*)take(nM1 * 4);
  unsigned char* m1u8 = (unsigned char*)take(nM1);
  float*  m2f     = (float*)take(nTE * 4);
  (void)ws_size; (void)n_in; (void)in_sizes; (void)out_size;

  dim3 blk(256);
  auto g1 = [](long long n) { return dim3((unsigned)((n + 255) / 256)); };
  const float invSqrtD = 0.125f;

  init_scales_kernel<<<1, 64, 0, stream>>>(scales);

  absmax_kernel<<<2048, blk, 0, stream>>>(x,  nTE,              scales + SL_X,  0);
  absmax_kernel<<<1024, blk, 0, stream>>>(Wq, (long long)EE*EE, scales + SL_WQ, 0);
  absmax_kernel<<<1024, blk, 0, stream>>>(Wk, (long long)EE*EE, scales + SL_WK, 0);
  absmax_kernel<<<1024, blk, 0, stream>>>(Wv, (long long)EE*EE, scales + SL_WV, 0);
  absmax_kernel<<<1024, blk, 0, stream>>>(Wo, (long long)EE*EE, scales + SL_WO, 0);
  absmax_kernel<<<2048, blk, 0, stream>>>(W1, (long long)MLPN*EE, scales + SL_W1, 0);
  absmax_kernel<<<2048, blk, 0, stream>>>(W2, (long long)EE*MLPN, scales + SL_W2, 0);
  absmax_kernel<<<8, blk, 0, stream>>>(bq, EE,   scales + SL_BQ, 0);
  absmax_kernel<<<8, blk, 0, stream>>>(bk, EE,   scales + SL_BK, 0);
  absmax_kernel<<<8, blk, 0, stream>>>(bv, EE,   scales + SL_BV, 0);
  absmax_kernel<<<8, blk, 0, stream>>>(bo, EE,   scales + SL_BO, 0);
  absmax_kernel<<<16, blk, 0, stream>>>(b1, MLPN, scales + SL_B1, 0);
  absmax_kernel<<<8, blk, 0, stream>>>(b2, EE,   scales + SL_B2, 0);

  quant_i8_kernel<<<g1(nTE), blk, 0, stream>>>(x, qx, scales, SL_X, 1.0f, -128.0f, 127.0f, nTE);
  quant_i8_kernel<<<g1((long long)EE*EE), blk, 0, stream>>>(Wq, qWq, scales, SL_WQ, 1.0f, -127.0f, 127.0f, (long long)EE*EE);
  quant_i8_kernel<<<g1((long long)EE*EE), blk, 0, stream>>>(Wk, qWk, scales, SL_WK, 1.0f, -127.0f, 127.0f, (long long)EE*EE);
  quant_i8_kernel<<<g1((long long)EE*EE), blk, 0, stream>>>(Wv, qWv, scales, SL_WV, 1.0f, -127.0f, 127.0f, (long long)EE*EE);
  quant_i8_kernel<<<g1((long long)EE*EE), blk, 0, stream>>>(Wo, qWo, scales, SL_WO, 1.0f, -127.0f, 127.0f, (long long)EE*EE);
  quant_i8_kernel<<<g1((long long)MLPN*EE), blk, 0, stream>>>(W1, qW1, scales, SL_W1, 1.0f, -127.0f, 127.0f, (long long)MLPN*EE);
  quant_i8_kernel<<<g1((long long)EE*MLPN), blk, 0, stream>>>(W2, qW2, scales, SL_W2, 1.0f, -127.0f, 127.0f, (long long)EE*MLPN);
  fq_float_kernel<<<g1(EE), blk, 0, stream>>>(bq, bqf, scales, SL_BQ, -128.0f, 127.0f, EE);
  fq_float_kernel<<<g1(EE), blk, 0, stream>>>(bk, bkf, scales, SL_BK, -128.0f, 127.0f, EE);
  fq_float_kernel<<<g1(EE), blk, 0, stream>>>(bv, bvf, scales, SL_BV, -128.0f, 127.0f, EE);
  fq_float_kernel<<<g1(EE), blk, 0, stream>>>(bo, bof, scales, SL_BO, -128.0f, 127.0f, EE);
  fq_float_kernel<<<g1(MLPN), blk, 0, stream>>>(b1, b1f, scales, SL_B1, -128.0f, 127.0f, MLPN);
  fq_float_kernel<<<g1(EE), blk, 0, stream>>>(b2, b2f, scales, SL_B2, -128.0f, 127.0f, EE);

  dim3 gProj(EE / 64, NT / 256, 1);
  gemm_i8_wmma<true, true><<<gProj, blk, 0, stream>>>(qx, qWq, qf, NT, EE, EE, 0, 0, 0,
      scales, SL_X, 127.0f, 1.0f, SL_WQ, bqf);
  gemm_i8_wmma<true, true><<<gProj, blk, 0, stream>>>(qx, qWk, kf, NT, EE, EE, 0, 0, 0,
      scales, SL_X, 127.0f, 1.0f, SL_WK, bkf);
  gemm_i8_wmma<true, true><<<gProj, blk, 0, stream>>>(qx, qWv, vf, NT, EE, EE, 0, 0, 0,
      scales, SL_X, 127.0f, 1.0f, SL_WV, bvf);

  absmax_kernel<<<2048, blk, 0, stream>>>(qf, nTE, scales + SL_Q, 0);
  absmax_kernel<<<2048, blk, 0, stream>>>(kf, nTE, scales + SL_K, 0);
  absmax_kernel<<<2048, blk, 0, stream>>>(vf, nTE, scales + SL_V, 0);
  quant_perm_kernel<<<g1(nTE), blk, 0, stream>>>(qf, q8,  scales, SL_Q, invSqrtD, 0);
  quant_perm_kernel<<<g1(nTE), blk, 0, stream>>>(kf, k8,  scales, SL_K, 1.0f,     0);
  quant_perm_kernel<<<g1(nTE), blk, 0, stream>>>(vf, v8t, scales, SL_V, 1.0f,     1);

  dim3 gScore(SS_ / 64, SS_ / 256, BB * HH);
  gemm_i8_wmma<true, true><<<gScore, blk, 0, stream>>>(q8, k8, scoresf, SS_, SS_, DD,
      (long long)SS_ * DD, (long long)SS_ * DD, (long long)SS_ * SS_,
      scales, SL_Q, 127.0f, invSqrtD, SL_K, nullptr);

  absmax_kernel<<<4096, blk, 0, stream>>>(scoresf, nSc, scales + SL_SC, 0);
  softmax_fq_kernel<<<BB * HH * SS_, blk, 0, stream>>>(scoresf, scales, SL_SC);
  absmax_kernel<<<4096, blk, 0, stream>>>(scoresf, nSc, scales + SL_AT, 0);
  quant_i8_kernel<<<g1(nSc), blk, 0, stream>>>(scoresf, attn8, scales, SL_AT, 1.0f, -128.0f, 127.0f, nSc);

  dim3 gCtx(DD / 64, SS_ / 256, BB * HH);
  gemm_i8_wmma<true, true><<<gCtx, blk, 0, stream>>>(attn8, v8t, ctxf, SS_, DD, SS_,
      (long long)SS_ * SS_, (long long)DD * SS_, (long long)SS_ * DD,
      scales, SL_AT, 127.0f, 1.0f, SL_V, nullptr);

  absmax_kernel<<<2048, blk, 0, stream>>>(ctxf, nTE, scales + SL_CTX, 0);
  quant_perm_kernel<<<g1(nTE), blk, 0, stream>>>(ctxf, ctx8, scales, SL_CTX, 1.0f, 2);
  gemm_i8_wmma<true, true><<<gProj, blk, 0, stream>>>(ctx8, qWo, aof, NT, EE, EE, 0, 0, 0,
      scales, SL_CTX, 127.0f, 1.0f, SL_WO, bof);

  absmax_kernel<<<2048, blk, 0, stream>>>(aof, nTE, scales + SL_AO, 0);
  add_fq_ln_kernel<<<NT, blk, 0, stream>>>(x, aof, scales, SL_X, SL_AO, hf);

  absmax_kernel<<<2048, blk, 0, stream>>>(hf, nTE, scales + SL_H, 0);
  quant_i8_kernel<<<g1(nTE), blk, 0, stream>>>(hf, h8, scales, SL_H, 1.0f, -128.0f, 127.0f, nTE);
  dim3 gM1(MLPN / 64, NT / 256, 1);
  gemm_i8_wmma<true, true><<<gM1, blk, 0, stream>>>(h8, qW1, m1f, NT, MLPN, EE, 0, 0, 0,
      scales, SL_H, 127.0f, 1.0f, SL_W1, b1f);
  absmax_kernel<<<4096, blk, 0, stream>>>(m1f, nM1, scales + SL_M1, 1  );
  relu_quant_u8_kernel<<<g1(nM1), blk, 0, stream>>>(m1f, m1u8, scales, SL_M1, nM1);
  dim3 gM2(EE / 64, NT / 256, 1);
  gemm_i8_wmma<false, true><<<gM2, blk, 0, stream>>>((const int8_t*)m1u8, qW2, m2f,
      NT, EE, MLPN, 0, 0, 0, scales, SL_M1, 255.0f, 1.0f, SL_W2, b2f);

  absmax_kernel<<<2048, blk, 0, stream>>>(m2f, nTE, scales + SL_M2, 0);
  add_fq_ln_kernel<<<NT, blk, 0, stream>>>(hf, m2f, scales, SL_H, SL_M2, out);
}
